// WriteHead_6399501271466
// MI455X (gfx1250) — matrix operand order repaired and checked by reading, not compiled, not run on hardware
//
#include <hip/hip_runtime.h>
#include <hip/hip_bf16.h>

#define B_   32
#define FPV  4
#define R_   8192
#define C_   128
#define HID_ 512
#define NH_  4
#define HD_  32
#define OUTW 390

typedef __attribute__((ext_vector_type(16))) __bf16 v16bf;
typedef __attribute__((ext_vector_type(8)))  float  v8f;

union BF16x16 { v16bf v; unsigned short u[16]; };

__device__ __forceinline__ unsigned short f2bf(float f) {
    unsigned int u = __float_as_uint(f);
    unsigned int r = u + 0x7FFFu + ((u >> 16) & 1u);   // round-to-nearest-even
    return (unsigned short)(r >> 16);
}
__device__ __forceinline__ float softplusf(float x) {
    return fmaxf(x, 0.0f) + log1pf(expf(-fabsf(x)));
}
__device__ __forceinline__ float sigmoidf_(float x) {
    return 1.0f / (1.0f + expf(-x));
}

// ---------------- Kernel 0: wq -> bf16 (padded stride, matches LDS copy) ----------------
__global__ void k_wq_bf16(const float* __restrict__ wq, unsigned short* __restrict__ wqb) {
    int i = blockIdx.x * blockDim.x + threadIdx.x;
    if (i < C_ * C_) {
        int n = i / C_, k = i % C_;
        wqb[n * 136 + k] = f2bf(wq[i]);
    }
}

// ---------------- Kernel 1: fc + split activations ----------------
__global__ void k_fc_act(const float* __restrict__ h, const float* __restrict__ fc_w,
                         const float* __restrict__ fc_b,
                         float* ks, float* es, float* as_,
                         float* beta, float* g, float* gamma, float* sraw) {
    int bf = blockIdx.x;
    __shared__ float hrow[HID_];
    for (int i = threadIdx.x; i < HID_; i += blockDim.x) hrow[i] = h[(size_t)bf * HID_ + i];
    __syncthreads();
    for (int j = threadIdx.x; j < OUTW; j += blockDim.x) {
        const float* wrow = fc_w + (size_t)j * HID_;
        float acc = fc_b[j];
        for (int k = 0; k < HID_; ++k) acc = fmaf(wrow[k], hrow[k], acc);
        if      (j < 128)  ks[bf * C_ + j] = tanhf(acc);
        else if (j == 128) beta[bf]  = softplusf(acc);
        else if (j == 129) g[bf]     = sigmoidf_(acc);
        else if (j < 133)  sraw[bf * 3 + (j - 130)] = acc;
        else if (j == 133) gamma[bf] = softplusf(acc) + 1.0f;
        else if (j < 262)  es[bf * C_ + (j - 134)] = sigmoidf_(acc);
        else               as_[bf * C_ + (j - 262)] = tanhf(acc);
    }
}

// ---------------- Kernel 2: kp = k @ wk^T + bk ; softmax of shift kernel ----------------
__global__ void k_kp(const float* __restrict__ ks, const float* __restrict__ wk,
                     const float* __restrict__ bk, const float* __restrict__ sraw,
                     float* kp, float* s) {
    int bf = blockIdx.x;
    __shared__ float krow[C_];
    int t = threadIdx.x;
    krow[t] = ks[bf * C_ + t];
    __syncthreads();
    const float* wrow = wk + (size_t)t * C_;
    float acc = bk[t];
    for (int k = 0; k < C_; ++k) acc = fmaf(wrow[k], krow[k], acc);
    kp[bf * C_ + t] = acc;
    if (t == 0) {
        float a0 = sraw[bf * 3], a1 = sraw[bf * 3 + 1], a2 = sraw[bf * 3 + 2];
        float m = fmaxf(a0, fmaxf(a1, a2));
        float e0 = expf(a0 - m), e1 = expf(a1 - m), e2 = expf(a2 - m);
        float inv = 1.0f / (e0 + e1 + e2);
        s[bf * 3] = e0 * inv; s[bf * 3 + 1] = e1 * inv; s[bf * 3 + 2] = e2 * inv;
    }
}

// ---------------- Kernel 3: WMMA q-GEMM fused with MHA scores ----------------
// q = mem_tile(16xC) @ wq^T (+bq) via v_wmma_f32_16x16x32_bf16, then
// scores[b,r,f] = mean_heads softmax_f( q[r,h*32:+32] . kp[f,h*32:+32] / sqrt(32) )
#define WQSTR 136   // padded ushort stride (16B-aligned rows, avoids bank conflicts)
#define QSTR  132   // padded float stride for staged q tile
__global__ void __launch_bounds__(64)
k_scores(const float* __restrict__ memory, const unsigned short* __restrict__ wqb,
         const float* __restrict__ bq, const float* __restrict__ kp,
         float* __restrict__ scores) {
    __shared__ unsigned short wqs[C_ * WQSTR];  // ~34 KB bf16 wq (B operand)
    __shared__ float kps[FPV * C_];             // 2 KB
    __shared__ float qls[2 * 16 * QSTR];        // ~17 KB, per-wave q staging

    const int b    = blockIdx.y;
    const int tid  = threadIdx.x;
    const int wave = tid >> 5;
    const int lane = tid & 31;
    const int m    = lane & 15;   // row within tile (A) / column within tile (B,C,D)
    const int hi   = lane >> 4;

    for (int i = tid; i < C_ * WQSTR; i += 64) wqs[i] = wqb[i];
    for (int i = tid; i < FPV * C_;  i += 64) kps[i] = kp[(size_t)b * FPV * C_ + i];
    __syncthreads();

    float* qbase = qls + wave * 16 * QSTR;

    for (int rt = 0; rt < 2; ++rt) {
        const int row0 = blockIdx.x * 64 + (wave * 2 + rt) * 16;
        const float4* arow4 = (const float4*)(memory + ((size_t)b * R_ + row0 + m) * C_);

        v8f acc[8];
        #pragma unroll
        for (int n = 0; n < 8; ++n) acc[n] = {};

        #pragma unroll
        for (int kstep = 0; kstep < 4; ++kstep) {
            const int kb = kstep * 32;
            // A fragment (16x32 bf16): lane<16 -> K {0..7,16..23}; lane>=16 -> K {8..15,24..31}
            // vector loads: 2x float4 at kb+hi*8, 2x float4 at kb+16+hi*8
            const int i0 = (kb + hi * 8) >> 2;
            float4 x0 = arow4[i0];
            float4 x1 = arow4[i0 + 1];
            float4 x2 = arow4[i0 + 4];
            float4 x3 = arow4[i0 + 5];
            v16bf av;   // native converts -> v_cvt_pk_bf16_f32
            av[0]  = (__bf16)x0.x; av[1]  = (__bf16)x0.y; av[2]  = (__bf16)x0.z; av[3]  = (__bf16)x0.w;
            av[4]  = (__bf16)x1.x; av[5]  = (__bf16)x1.y; av[6]  = (__bf16)x1.z; av[7]  = (__bf16)x1.w;
            av[8]  = (__bf16)x2.x; av[9]  = (__bf16)x2.y; av[10] = (__bf16)x2.z; av[11] = (__bf16)x2.w;
            av[12] = (__bf16)x3.x; av[13] = (__bf16)x3.y; av[14] = (__bf16)x3.z; av[15] = (__bf16)x3.w;
            #pragma unroll
            for (int n = 0; n < 8; ++n) {
                // B fragment (32x16 bf16): lane = column n*16+m; lane<16 -> K {0..7,16..23}; lane>=16 -> K {8..15,24..31}
                BF16x16 bfr;
                const unsigned short* bsrc = wqs + (n * 16 + m) * WQSTR + kb + hi * 8;
                #pragma unroll
                for (int t2 = 0; t2 < 16; ++t2) bfr.u[t2] = bsrc[t2 + (t2 >= 8 ? 8 : 0)];
                acc[n] = __builtin_amdgcn_wmma_f32_16x16x32_bf16(
                    false, av, false, bfr.v, (short)0, acc[n], false, false);
            }
        }
        // add bq, dump q tile to LDS (C/D layout: N = lane&15, M = j + 8*hi)
        #pragma unroll
        for (int n = 0; n < 8; ++n) {
            float bqv = bq[n * 16 + m];
            #pragma unroll
            for (int j = 0; j < 8; ++j)
                qbase[(j + hi * 8) * QSTR + n * 16 + m] = acc[n][j] + bqv;
        }
        __syncthreads();

        // per-lane: row = m; heads {2*hi, 2*hi+1}; softmax over F=4, mean over heads
        float p0 = 0.f, p1 = 0.f, p2 = 0.f, p3 = 0.f;
        #pragma unroll
        for (int hh = 0; hh < 2; ++hh) {
            int nh = 2 * hi + hh;
            float lg[4];
            #pragma unroll
            for (int f = 0; f < 4; ++f) {
                float dot = 0.f;
                const float* qrow  = qbase + m * QSTR + nh * HD_;
                const float* krow2 = kps + f * C_ + nh * HD_;
                for (int d = 0; d < HD_; ++d) dot = fmaf(qrow[d], krow2[d], dot);
                lg[f] = dot * 0.17677669529663687f;   // 1/sqrt(32)
            }
            float mx = fmaxf(fmaxf(lg[0], lg[1]), fmaxf(lg[2], lg[3]));
            float e0 = expf(lg[0] - mx), e1 = expf(lg[1] - mx);
            float e2 = expf(lg[2] - mx), e3 = expf(lg[3] - mx);
            float inv = 0.25f / (e0 + e1 + e2 + e3);   // mean over NH=4 heads
            p0 += e0 * inv; p1 += e1 * inv; p2 += e2 * inv; p3 += e3 * inv;
        }
        p0 += __shfl_xor(p0, 16, 32);
        p1 += __shfl_xor(p1, 16, 32);
        p2 += __shfl_xor(p2, 16, 32);
        p3 += __shfl_xor(p3, 16, 32);
        if (hi == 0) {
            *(float4*)(scores + ((size_t)b * R_ + row0 + m) * 4) = make_float4(p0, p1, p2, p3);
        }
        __syncthreads();
    }
}

// ---------------- Kernel 4: softmax over R (w_c) + gated interpolation ----------------
__global__ void k_softmax_gate(const float* __restrict__ scores, const float* __restrict__ beta,
                               const float* __restrict__ g, const float* __restrict__ prev_w,
                               float* __restrict__ wg) {
    int bf = blockIdx.x; int b = bf >> 2; int f = bf & 3;
    float bv = beta[bf], gv = g[bf];
    __shared__ float red[256];
    int t = threadIdx.x;
    const float* sc = scores + (size_t)b * R_ * 4 + f;
    float mx = -3.4e38f;
    for (int r = t; r < R_; r += 256) mx = fmaxf(mx, bv * sc[(size_t)r * 4]);
    red[t] = mx; __syncthreads();
    for (int s2 = 128; s2 > 0; s2 >>= 1) { if (t < s2) red[t] = fmaxf(red[t], red[t + s2]); __syncthreads(); }
    mx = red[0]; __syncthreads();
    float sum = 0.f;
    for (int r = t; r < R_; r += 256) sum += expf(bv * sc[(size_t)r * 4] - mx);
    red[t] = sum; __syncthreads();
    for (int s2 = 128; s2 > 0; s2 >>= 1) { if (t < s2) red[t] += red[t + s2]; __syncthreads(); }
    float inv = 1.0f / red[0];
    const float* pw  = prev_w + (size_t)bf * R_;
    float*       wgo = wg     + (size_t)bf * R_;
    for (int r = t; r < R_; r += 256) {
        float wc = expf(bv * sc[(size_t)r * 4] - mx) * inv;
        wgo[r] = gv * wc + (1.0f - gv) * pw[r];
    }
}

// ---------------- Kernel 5: circular shift + sharpen + normalize -> w ----------------
__global__ void k_shift_sharpen(const float* __restrict__ wg, const float* __restrict__ s,
                                const float* __restrict__ gamma,
                                float* __restrict__ wp, float* __restrict__ wout) {
    int bf = blockIdx.x;
    float s0 = s[bf * 3], s1 = s[bf * 3 + 1], s2 = s[bf * 3 + 2], gm = gamma[bf];
    const float* wgi = wg + (size_t)bf * R_;
    float*       wpo = wp + (size_t)bf * R_;
    __shared__ float red[256];
    int t = threadIdx.x;
    float sum = 0.f;
    for (int r = t; r < R_; r += 256) {
        int rm = (r == 0)      ? (R_ - 1) : r - 1;
        int rp = (r == R_ - 1) ? 0        : r + 1;
        float v = s0 * wgi[rm] + s1 * wgi[r] + s2 * wgi[rp];
        float p = powf(v, gm);
        wpo[r] = p; sum += p;
    }
    red[t] = sum; __syncthreads();
    for (int s3 = 128; s3 > 0; s3 >>= 1) { if (t < s3) red[t] += red[t + s3]; __syncthreads(); }
    float inv = 1.0f / (red[0] + 1e-16f);
    float* wo = wout + (size_t)bf * R_;
    for (int r = t; r < R_; r += 256) wo[r] = wpo[r] * inv;
}

// ---------------- Kernel 6: erase/add memory write ----------------
__global__ void k_memwrite(const float* __restrict__ memory, const float* __restrict__ w,
                           const float* __restrict__ es, const float* __restrict__ as_,
                           float* __restrict__ newmem) {
    int b = blockIdx.y;
    int t = threadIdx.x;
    int r = blockIdx.x * 2 + (t >> 7);
    int c = t & 127;
    float w0 = w[((size_t)b * 4 + 0) * R_ + r];
    float w1 = w[((size_t)b * 4 + 1) * R_ + r];
    float w2 = w[((size_t)b * 4 + 2) * R_ + r];
    float w3 = w[((size_t)b * 4 + 3) * R_ + r];
    int eb = b * 4 * C_ + c;
    float we = w0 * es[eb] + w1 * es[eb + C_] + w2 * es[eb + 2 * C_] + w3 * es[eb + 3 * C_];
    float wa = w0 * as_[eb] + w1 * as_[eb + C_] + w2 * as_[eb + 2 * C_] + w3 * as_[eb + 3 * C_];
    size_t idx = ((size_t)b * R_ + r) * C_ + c;
    newmem[idx] = fmaf(memory[idx], we, wa);
}

extern "C" void kernel_launch(void* const* d_in, const int* in_sizes, int n_in,
                              void* d_out, int out_size, void* d_ws, size_t ws_size,
                              hipStream_t stream) {
    const float* h      = (const float*)d_in[0];
    const float* prev_w = (const float*)d_in[1];
    const float* memory = (const float*)d_in[2];
    const float* fc_w   = (const float*)d_in[3];
    const float* fc_b   = (const float*)d_in[4];
    const float* wq     = (const float*)d_in[5];
    const float* wk     = (const float*)d_in[6];
    const float* bq     = (const float*)d_in[7];
    const float* bk     = (const float*)d_in[8];

    // workspace layout (floats)
    float* F = (float*)d_ws;
    float* ks     = F;                 // 16384
    float* es     = F + 16384;         // 16384
    float* as_    = F + 32768;         // 16384
    float* kp     = F + 49152;         // 16384
    float* beta   = F + 65536;         // 128
    float* g      = F + 65664;         // 128
    float* gamma  = F + 65792;         // 128
    float* sraw   = F + 65920;         // 384
    float* s      = F + 66304;         // 384
    float* scores = F + 66688;         // B*R*F = 1048576
    float* wg     = scores + 1048576;  // 1048576
    float* wp     = wg + 1048576;      // 1048576
    unsigned short* wqb = (unsigned short*)(wp + 1048576);  // 128*136 ushorts

    float* wout   = (float*)d_out;                      // [B,F,R]
    float* newmem = wout + (size_t)B_ * FPV * R_;       // [B,R,C]

    k_wq_bf16<<<(C_ * C_ + 255) / 256, 256, 0, stream>>>(wq, wqb);
    k_fc_act<<<B_ * FPV, 128, 0, stream>>>(h, fc_w, fc_b, ks, es, as_, beta, g, gamma, sraw);
    k_kp<<<B_ * FPV, 128, 0, stream>>>(ks, wk, bk, sraw, kp, s);
    k_scores<<<dim3(R_ / 64, B_), 64, 0, stream>>>(memory, wqb, bq, kp, scores);
    k_softmax_gate<<<B_ * FPV, 256, 0, stream>>>(scores, beta, g, prev_w, wg);
    k_shift_sharpen<<<B_ * FPV, 256, 0, stream>>>(wg, s, gamma, wp, wout);
    k_memwrite<<<dim3(R_ / 2, B_), 256, 0, stream>>>(memory, wout, es, as_, newmem);
}
